// AttentionSequencePoolingLayer_18141941859094
// MI455X (gfx1250) — compile-verified, matrix operand order checked by reading, not run on hardware
//
#include <hip/hip_runtime.h>
#include <hip/hip_bf16.h>
#include <math.h>

typedef _Float16 v8h  __attribute__((ext_vector_type(8)));
typedef _Float16 v16h __attribute__((ext_vector_type(16)));
typedef float    v8f  __attribute__((ext_vector_type(8)));

#define B_   1024
#define T_   200
#define E_   64
#define H1_  64
#define H2_  16
#define NROW (B_*T_)          /* 204800 rows of (b,t) */
#define EPSF 1e-8f

// ---------------------------------------------------------------------------
// dice activation given precomputed channel mean / rstd / alpha
// ---------------------------------------------------------------------------
__device__ __forceinline__ float dice_f(float x, float mean, float rstd, float alpha) {
    float xn = (x - mean) * rstd;
    float p  = 1.0f / (1.0f + __expf(-xn));
    return p * x + (1.0f - p) * alpha * x;
}

// Load a 16-half WMMA fragment: two contiguous 8-half (16B) runs, 16 halfs apart.
// Works for LDS (ds_load_b128) and global (global_load_b128) pointers.
__device__ __forceinline__ v16h ld_frag(const _Float16* p) {
    v8h lo = *(const v8h*)(p);
    v8h hi = *(const v8h*)(p + 16);
    return __builtin_shufflevector(lo, hi, 0,1,2,3,4,5,6,7,8,9,10,11,12,13,14,15);
}

// ---------------------------------------------------------------------------
// Kernel 0: zero the stats region; convert W1 -> f16 transposed [n][k],
//           W2 -> f16 transposed [n][k] (resident in L2, reused by all blocks)
// ---------------------------------------------------------------------------
__global__ __launch_bounds__(256)
void k0_init(const float* __restrict__ W1, const float* __restrict__ W2,
             float* __restrict__ stats, _Float16* __restrict__ W1t,
             _Float16* __restrict__ W2t) {
    const int tid = threadIdx.x;
    if (tid < 192) stats[tid] = 0.0f;
    for (int idx = tid; idx < 256 * 64; idx += 256) {
        int n = idx & 63, k = idx >> 6;            // W1 row-major (k, n)
        W1t[n * 256 + k] = (_Float16)W1[idx];
    }
    for (int idx = tid; idx < 64 * 16; idx += 256) {
        int n = idx & 15, k = idx >> 4;            // W2 row-major (k, n)
        W2t[n * 64 + k] = (_Float16)W2[idx];
    }
}

// ---------------------------------------------------------------------------
// Kernel 1: z1 = att_in @ W1 + b1   (M=204800, K=256, N=64) in f16 WMMA.
// att_in = [q, ub, q-ub, q*ub] synthesized on the fly into LDS.
// Also accumulates per-channel sum / sumsq for the first dice().
// Block = 256 threads = 8 waves; each wave: 16 rows x 64 cols.
// ---------------------------------------------------------------------------
__global__ __launch_bounds__(256)
void k1_gemm1(const float* __restrict__ q, const float* __restrict__ ub,
              const _Float16* __restrict__ W1t, const float* __restrict__ b1,
              _Float16* __restrict__ z1h, float* __restrict__ gSum,
              float* __restrict__ gSsq) {
    __shared__ _Float16 sA[128 * 256];             // 64KB att_in tile (f16)
    const int tid  = threadIdx.x;
    const int row0 = blockIdx.x * 128;

    // ---- stage att_in tile: thread (rg, e) covers rows rg+4j, column e ----
    const int e  = tid & 63;
    const int rg = tid >> 6;
    for (int j = 0; j < 32; ++j) {
        int r   = rg + 4 * j;
        int row = row0 + r;
        int b   = row / T_;
        float qv = q[b * E_ + e];
        float uv = ub[(size_t)row * E_ + e];       // row == b*T + t
        _Float16* pr = &sA[r * 256];
        pr[e]        = (_Float16)qv;
        pr[64  + e]  = (_Float16)uv;
        pr[128 + e]  = (_Float16)(qv - uv);
        pr[192 + e]  = (_Float16)(qv * uv);
    }
    __syncthreads();

    const int wave = tid >> 5;
    const int lane = tid & 31;
    const int mloc = lane & 15;
    const int kb   = (lane >> 4) * 8;              // ISA 16-bit A/B lane K-offset
    const int rloc = wave * 16 + mloc;

    v8f c0 = {}, c1 = {}, c2 = {}, c3 = {};
    for (int k0 = 0; k0 < 256; k0 += 32) {
        v16h a  = ld_frag(&sA[rloc * 256 + k0 + kb]);
        v16h b0 = ld_frag(&W1t[( 0 + mloc) * 256 + k0 + kb]);
        v16h bf1 = ld_frag(&W1t[(16 + mloc) * 256 + k0 + kb]);
        v16h bf2 = ld_frag(&W1t[(32 + mloc) * 256 + k0 + kb]);
        v16h bf3 = ld_frag(&W1t[(48 + mloc) * 256 + k0 + kb]);
        c0 = __builtin_amdgcn_wmma_f32_16x16x32_f16(false, a, false, b0,  (short)0, c0, false, false);
        c1 = __builtin_amdgcn_wmma_f32_16x16x32_f16(false, a, false, bf1, (short)0, c1, false, false);
        c2 = __builtin_amdgcn_wmma_f32_16x16x32_f16(false, a, false, bf2, (short)0, c2, false, false);
        c3 = __builtin_amdgcn_wmma_f32_16x16x32_f16(false, a, false, bf3, (short)0, c3, false, false);
    }

    // ---- epilogue: +b1, store z1 (f16), channel stats ----
    const int mbase = row0 + wave * 16 + ((lane >> 4) * 8);  // C layout: M = r + 8*(lane>=16)
    v8f cs[4] = {c0, c1, c2, c3};
    for (int nt = 0; nt < 4; ++nt) {
        int n = nt * 16 + mloc;                    // C layout: N = lane&15
        float bias = b1[n];
        float ps = 0.0f, pq = 0.0f;
        v8f cc = cs[nt];
        for (int r = 0; r < 8; ++r) {
            float v = cc[r] + bias;
            z1h[(size_t)(mbase + r) * H1_ + n] = (_Float16)v;
            ps += v; pq += v * v;
        }
        // lanes l and l+16 share the same column n -> reduce across half-waves
        ps += __shfl_xor(ps, 16);
        pq += __shfl_xor(pq, 16);
        if (lane < 16) {
            atomicAdd(&gSum[n], ps);
            atomicAdd(&gSsq[n], pq);
        }
    }
}

// ---------------------------------------------------------------------------
// Kernel 2: h1 = dice(z1); z2 = h1 @ W2 + b2  (K=64, N=16) in f16 WMMA.
// Finalizes dice-1 stats from global sums; accumulates dice-2 stats.
// ---------------------------------------------------------------------------
__global__ __launch_bounds__(256)
void k2_gemm2(const _Float16* __restrict__ z1h, const float* __restrict__ a1,
              const _Float16* __restrict__ W2t, const float* __restrict__ b2,
              const float* __restrict__ gSum1, const float* __restrict__ gSsq1,
              _Float16* __restrict__ z2h, float* __restrict__ gSum2,
              float* __restrict__ gSsq2) {
    __shared__ _Float16 sH[128 * 64];              // 16KB dice(z1) tile (f16)
    __shared__ float sSum[16], sSsq[16];
    const int tid  = threadIdx.x;
    const int row0 = blockIdx.x * 128;
    const float Nf = (float)NROW;

    // per-thread channel stats (column cch)
    const int cch = tid & 63;
    float S = gSum1[cch], SS = gSsq1[cch];
    float mean = S / Nf;
    float var  = SS / Nf - mean * mean;
    float rstd = rsqrtf(var + EPSF);
    float alpha = a1[cch];

    const int rg = tid >> 6;
    for (int j = 0; j < 32; ++j) {
        int r = rg + 4 * j;
        float x = (float)z1h[(size_t)(row0 + r) * H1_ + cch];
        sH[r * 64 + cch] = (_Float16)dice_f(x, mean, rstd, alpha);
    }
    if (tid < 16) { sSum[tid] = 0.0f; sSsq[tid] = 0.0f; }
    __syncthreads();

    const int wave = tid >> 5, lane = tid & 31;
    const int mloc = lane & 15, kb = (lane >> 4) * 8;
    const int rloc = wave * 16 + mloc;
    v8f c = {};
    for (int k0 = 0; k0 < 64; k0 += 32) {
        v16h a  = ld_frag(&sH[rloc * 64 + k0 + kb]);
        v16h bf = ld_frag(&W2t[mloc * 64 + k0 + kb]);
        c = __builtin_amdgcn_wmma_f32_16x16x32_f16(false, a, false, bf, (short)0, c, false, false);
    }
    const int mbase = row0 + wave * 16 + ((lane >> 4) * 8);
    const int n = mloc;
    float bias = b2[n];
    float ps = 0.0f, pq = 0.0f;
    for (int r = 0; r < 8; ++r) {
        float v = c[r] + bias;
        z2h[(size_t)(mbase + r) * H2_ + n] = (_Float16)v;
        ps += v; pq += v * v;
    }
    atomicAdd(&sSum[n], ps);
    atomicAdd(&sSsq[n], pq);
    __syncthreads();
    if (tid < 16) {
        atomicAdd(&gSum2[tid], sSum[tid]);
        atomicAdd(&gSsq2[tid], sSsq[tid]);
    }
}

// ---------------------------------------------------------------------------
// Kernel 3: score = dice(z2) @ W3 + b3 (length-16 dot, one thread per row).
// Accumulates global sum/sumsq of score for the final (scalar) dice.
// ---------------------------------------------------------------------------
__global__ __launch_bounds__(256)
void k3_score(const _Float16* __restrict__ z2h, const float* __restrict__ a2,
              const float* __restrict__ W3, const float* __restrict__ b3,
              const float* __restrict__ gSum2, const float* __restrict__ gSsq2,
              float* __restrict__ score, float* __restrict__ gSum3,
              float* __restrict__ gSsq3) {
    __shared__ float m2[16], rs2[16], al2[16], w3s[16];
    __shared__ float bpair[2];
    const int tid = threadIdx.x;
    if (tid < 16) {
        float Nf = (float)NROW;
        float mean = gSum2[tid] / Nf;
        float var  = gSsq2[tid] / Nf - mean * mean;
        m2[tid]  = mean;
        rs2[tid] = rsqrtf(var + EPSF);
        al2[tid] = a2[tid];
        w3s[tid] = W3[tid];
    }
    if (tid == 0) { bpair[0] = 0.0f; bpair[1] = 0.0f; }
    __syncthreads();

    const int row = blockIdx.x * 256 + tid;
    float s = b3[0];
    #pragma unroll
    for (int i = 0; i < 16; ++i) {
        float x = (float)z2h[(size_t)row * H2_ + i];
        s += dice_f(x, m2[i], rs2[i], al2[i]) * w3s[i];
    }
    score[row] = s;
    atomicAdd(&bpair[0], s);
    atomicAdd(&bpair[1], s * s);
    __syncthreads();
    if (tid == 0) { atomicAdd(gSum3, bpair[0]); atomicAdd(gSsq3, bpair[1]); }
}

// ---------------------------------------------------------------------------
// Kernel 4: final dice on score, length-mask, softmax over T, and
// out[b] = sum_t w_t * ub[b,t,:].  One block per batch row.
// ---------------------------------------------------------------------------
__global__ __launch_bounds__(256)
void k4_softmax_pool(const float* __restrict__ score, const float* __restrict__ ub,
                     const long long* __restrict__ lens, const float* __restrict__ a3,
                     const float* __restrict__ gSum3, const float* __restrict__ gSsq3,
                     float* __restrict__ out) {
    __shared__ float sw[T_];
    __shared__ float red[256];
    __shared__ float acc4[4][64];
    const int tid = threadIdx.x;
    const int b   = blockIdx.x;
    const float Nf = (float)NROW;
    float mean  = gSum3[0] / Nf;
    float var   = gSsq3[0] / Nf - mean * mean;
    float rstd  = rsqrtf(var + EPSF);
    float alpha = a3[0];
    const int len = (int)lens[b];

    float mv = -__builtin_inff();
    if (tid < T_) {
        float x = score[b * T_ + tid];
        float y = dice_f(x, mean, rstd, alpha);
        mv = (tid < len) ? y : -__builtin_inff();
        sw[tid] = mv;
    }
    red[tid] = mv; __syncthreads();
    for (int s = 128; s > 0; s >>= 1) {
        if (tid < s) red[tid] = fmaxf(red[tid], red[tid + s]);
        __syncthreads();
    }
    float mx = red[0];
    __syncthreads();
    float p = 0.0f;
    if (tid < T_) p = __expf(sw[tid] - mx);        // exp(-inf)=0 handles mask
    red[tid] = p; __syncthreads();
    for (int s = 128; s > 0; s >>= 1) {
        if (tid < s) red[tid] += red[tid + s];
        __syncthreads();
    }
    float inv = 1.0f / red[0];
    __syncthreads();
    if (tid < T_) sw[tid] = p * inv;
    __syncthreads();

    // weighted pooling: thread (ch, e) sums t = ch, ch+4, ...
    const int e = tid & 63, ch = tid >> 6;
    float acc = 0.0f;
    for (int t = ch; t < T_; t += 4) {
        if (t + 4 < T_)
            __builtin_prefetch(&ub[((size_t)b * T_ + t + 4) * E_ + e], 0, 1);
        acc += sw[t] * ub[((size_t)b * T_ + t) * E_ + e];
    }
    acc4[ch][e] = acc;
    __syncthreads();
    if (tid < 64)
        out[b * E_ + tid] = acc4[0][tid] + acc4[1][tid] + acc4[2][tid] + acc4[3][tid];
}

// ---------------------------------------------------------------------------
extern "C" void kernel_launch(void* const* d_in, const int* in_sizes, int n_in,
                              void* d_out, int out_size, void* d_ws, size_t ws_size,
                              hipStream_t stream) {
    const float*     q    = (const float*)d_in[0];
    const float*     ub   = (const float*)d_in[1];
    const long long* lens = (const long long*)d_in[2];   // int64 per reference
    const float* W1 = (const float*)d_in[3];
    const float* b1 = (const float*)d_in[4];
    const float* a1 = (const float*)d_in[5];
    const float* W2 = (const float*)d_in[6];
    const float* b2 = (const float*)d_in[7];
    const float* a2 = (const float*)d_in[8];
    const float* W3 = (const float*)d_in[9];
    const float* b3 = (const float*)d_in[10];
    const float* a3 = (const float*)d_in[11];
    float* out = (float*)d_out;

    // workspace layout (all offsets 16B aligned)
    _Float16* z1h   = (_Float16*)d_ws;                       // NROW*64 halfs
    _Float16* z2h   = z1h + (size_t)NROW * H1_;              // NROW*16 halfs
    float*    score = (float*)(z2h + (size_t)NROW * H2_);    // NROW floats
    float*    stats = score + NROW;                          // 192 floats
    float* gSum1 = stats;        float* gSsq1 = stats + 64;
    float* gSum2 = stats + 128;  float* gSsq2 = stats + 144;
    float* gSum3 = stats + 160;  float* gSsq3 = stats + 161;
    _Float16* W1t = (_Float16*)(stats + 192);                // 64*256 halfs
    _Float16* W2t = W1t + 64 * 256;                          // 16*64 halfs

    k0_init<<<1, 256, 0, stream>>>(W1, W2, stats, W1t, W2t);
    k1_gemm1<<<NROW / 128, 256, 0, stream>>>(q, ub, W1t, b1, z1h, gSum1, gSsq1);
    k2_gemm2<<<NROW / 128, 256, 0, stream>>>(z1h, a1, W2t, b2, gSum1, gSsq1,
                                             z2h, gSum2, gSsq2);
    k3_score<<<NROW / 256, 256, 0, stream>>>(z2h, a2, W3, b3, gSum2, gSsq2,
                                             score, gSum3, gSsq3);
    k4_softmax_pool<<<B_, 256, 0, stream>>>(score, ub, lens, a3, gSum3, gSsq3, out);
}
